// SwitchSAE_71150428225656
// MI455X (gfx1250) — matrix operand order repaired and checked by reading, not compiled, not run on hardware
//
#include <hip/hip_runtime.h>

#define Nn 4096
#define Hh 1024
#define Ee 16
#define Ff 1024
#define AST 1032  // LDS A row stride in halfs: 16B-aligned, spreads banks

typedef __attribute__((ext_vector_type(16))) __bf16 v16bf;
typedef __attribute__((ext_vector_type(8)))  float  v8f;

union Frag { uint4 q[2]; v16bf v; };

__device__ __forceinline__ unsigned short f2bf(float x) {
  unsigned u = __float_as_uint(x);
  u += 0x7FFFu + ((u >> 16) & 1u);   // round-to-nearest-even
  return (unsigned short)(u >> 16);
}

// CDNA5 async copy global->LDS (16B per lane), tracked by ASYNCcnt.
__device__ __forceinline__ void async_b128(unsigned lds_off, const void* g) {
  asm volatile("global_load_async_to_lds_b128 %0, %1, off"
               :: "v"(lds_off), "v"(g) : "memory");
}
__device__ __forceinline__ void wait_async0() {
  asm volatile("s_wait_asynccnt 0x0" ::: "memory");
}
__device__ __forceinline__ void wait_async8() {
  asm volatile("s_wait_asynccnt 0x8" ::: "memory");
}

// ---- transpose f32 [R][C] -> bf16 [C][R], slice = blockIdx.z; block (32,8) ----
__global__ __launch_bounds__(256) void k_transpose(const float* __restrict__ src,
                                                   unsigned short* __restrict__ dst,
                                                   int R, int C) {
  __shared__ float tile[32][33];
  size_t slice = (size_t)blockIdx.z * R * C;
  const float* s = src + slice;
  unsigned short* d = dst + slice;
  int c0 = blockIdx.x * 32, r0 = blockIdx.y * 32;
  int tx = threadIdx.x, ty = threadIdx.y;
#pragma unroll
  for (int i = 0; i < 32; i += 8)
    tile[ty + i][tx] = s[(size_t)(r0 + ty + i) * C + (c0 + tx)];
  __syncthreads();
#pragma unroll
  for (int i = 0; i < 32; i += 8)
    d[(size_t)(c0 + ty + i) * R + (r0 + tx)] = f2bf(tile[tx][ty + i]);
}

// ---- router: logits, softmax max-prob, argmax, append token to expert list ----
__global__ __launch_bounds__(256) void k_router(const float* __restrict__ act,
                                                const float* __restrict__ router,
                                                const float* __restrict__ router_b,
                                                int* __restrict__ counts,
                                                int* __restrict__ token_list,
                                                float* __restrict__ max_prob) {
  int wave = threadIdx.x >> 5, lane = threadIdx.x & 31;
  int tbase = blockIdx.x * 64 + wave * 8;
  for (int i = 0; i < 8; i++) {
    int t = tbase + i;
    float acc[Ee];
#pragma unroll
    for (int e = 0; e < Ee; e++) acc[e] = 0.f;
    for (int h = lane; h < Hh; h += 32) {
      float x = act[(size_t)t * Hh + h] - router_b[h];
#pragma unroll
      for (int e = 0; e < Ee; e++) acc[e] += x * router[h * Ee + e];
    }
#pragma unroll
    for (int e = 0; e < Ee; e++) {
      float v = acc[e];
      for (int off = 16; off > 0; off >>= 1) v += __shfl_xor(v, off, 32);
      acc[e] = v;
    }
    if (lane == 0) {
      float m = acc[0]; int idx = 0;
#pragma unroll
      for (int e = 1; e < Ee; e++) if (acc[e] > m) { m = acc[e]; idx = e; }
      float s = 0.f;
#pragma unroll
      for (int e = 0; e < Ee; e++) s += __expf(acc[e] - m);
      max_prob[t] = 1.0f / s;              // softmax peak = 1/sum(exp(l-lmax))
      int pos = atomicAdd(&counts[idx], 1);
      token_list[idx * Nn + pos] = t;
    }
  }
}

// ---- fused routed encode(+relu) -> decode; 32 tokens x 1024 cols per block ----
// 8 waves; each wave: 2 row-tiles x 4 col-tiles per pass, 2 col passes.
// B tiles are double-buffered in LDS via async-to-LDS, laid out per-lane so
// ds_load_b128 reads back the exact WMMA B fragment.
__global__ __launch_bounds__(256, 1) void k_moe(const float* __restrict__ act,
                                                const float* __restrict__ expert_b,
                                                const unsigned short* __restrict__ encT,
                                                const unsigned short* __restrict__ decT,
                                                unsigned short* __restrict__ featbuf,
                                                const int* __restrict__ counts,
                                                const int* __restrict__ token_list,
                                                const float* __restrict__ max_prob,
                                                float* __restrict__ out) {
  int e = blockIdx.y, m = blockIdx.x;
  int cnt = counts[e];
  if (m * 32 >= cnt) return;

  __shared__ __align__(16) unsigned short Asm[32 * AST];      // 66 KB: xc rows bf16
  __shared__ __align__(16) unsigned char  Bsm[8 * 8192];      // 64 KB: per-wave 2x4KB
  __shared__ int   toks[32];
  __shared__ int   vld[32];
  __shared__ float mps[32];

  int tid = threadIdx.x;
  if (tid < 32) {
    int idx = m * 32 + tid;
    int v = idx < cnt;
    if (!v) idx = cnt - 1;                 // pad last tile with duplicate row
    int t = token_list[e * Nn + idx];
    toks[tid] = t; vld[tid] = v; mps[tid] = max_prob[t];
  }
  __syncthreads();

  // stage xc = act - expert_b as bf16 into LDS (32 rows x 1024)
  {
    int r = tid >> 3, seg = tid & 7;
    const float* arow = act + (size_t)toks[r] * Hh;
    unsigned short* srow = Asm + r * AST;
#pragma unroll 4
    for (int i = 0; i < 32; i++) {
      int c = seg * 128 + i * 4;
      float4 a = *(const float4*)(arow + c);
      float4 b = *(const float4*)(expert_b + c);
      unsigned lo = (unsigned)f2bf(a.x - b.x) | ((unsigned)f2bf(a.y - b.y) << 16);
      unsigned hi = (unsigned)f2bf(a.z - b.z) | ((unsigned)f2bf(a.w - b.w) << 16);
      *(uint2*)(srow + c) = make_uint2(lo, hi);
    }
  }
  __syncthreads();

  int wave = tid >> 5, lane = tid & 31, ln = lane & 15, kh = lane >> 4;
  // per-lane async destination slot (byte offset in LDS)
  unsigned bslot = (unsigned)(unsigned long long)(Bsm) + wave * 8192u + lane * 32u;
  const unsigned char* bread = Bsm + wave * 8192 + lane * 32;

  // ---------------- encode: feat = relu(xc @ enc[e]) ----------------
  const unsigned short* B0 = encT + (size_t)e * Ff * Hh;
#pragma unroll 1
  for (int cg = 0; cg < 2; cg++) {
    int colb = cg * 512 + wave * 64;
    v8f acc[2][4];
#pragma unroll
    for (int rt = 0; rt < 2; rt++)
#pragma unroll
      for (int t = 0; t < 4; t++) { v8f z; for (int j = 0; j < 8; j++) z[j] = 0.f; acc[rt][t] = z; }

    wait_async0();  // retire any leftover prefetches into our B buffers
#pragma unroll
    for (int t = 0; t < 4; t++) {          // prefetch k=0 into buffer 0
      const unsigned short* g = B0 + (((size_t)(colb + t * 16 + ln)) << 10) + kh * 16;
      async_b128(bslot + t * 1024u, g);
      async_b128(bslot + t * 1024u + 16u, g + 8);
    }
    unsigned p = 0;
#pragma unroll 1
    for (int k = 0; k < Hh; k += 32) {
      int kn = (k + 32 < Hh) ? k + 32 : k; // dummy prefetch keeps count uniform
      unsigned pn = p ^ 1u;
#pragma unroll
      for (int t = 0; t < 4; t++) {
        const unsigned short* g = B0 + (((size_t)(colb + t * 16 + ln)) << 10) + kn + kh * 16;
        async_b128(bslot + pn * 4096u + t * 1024u, g);
        async_b128(bslot + pn * 4096u + t * 1024u + 16u, g + 8);
      }
      wait_async8();                       // current buffer p is ready
      Frag a0, a1;
      const unsigned short* ar0 = Asm + (size_t)ln * AST + k + kh * 16;
      const unsigned short* ar1 = Asm + (size_t)(16 + ln) * AST + k + kh * 16;
      a0.q[0] = *(const uint4*)(ar0); a0.q[1] = *(const uint4*)(ar0 + 8);
      a1.q[0] = *(const uint4*)(ar1); a1.q[1] = *(const uint4*)(ar1 + 8);
#pragma unroll
      for (int t = 0; t < 4; t++) {
        Frag b;
        const unsigned char* bp = bread + p * 4096u + t * 1024u;
        b.q[0] = *(const uint4*)(bp);
        b.q[1] = *(const uint4*)(bp + 16);
        acc[0][t] = __builtin_amdgcn_wmma_f32_16x16x32_bf16(
            false, a0.v, false, b.v, (short)0, acc[0][t], false, false);
        acc[1][t] = __builtin_amdgcn_wmma_f32_16x16x32_bf16(
            false, a1.v, false, b.v, (short)0, acc[1][t], false, false);
      }
      p = pn;
    }
    // relu + bf16 -> feat scratch (by token id; C layout: vgpr j -> M = j + kh*8)
#pragma unroll
    for (int rt = 0; rt < 2; rt++)
#pragma unroll
      for (int t = 0; t < 4; t++) {
        int col = colb + t * 16 + ln;
#pragma unroll
        for (int j = 0; j < 8; j++) {
          int row = rt * 16 + j + kh * 8;
          float v = acc[rt][t][j];
          featbuf[(size_t)toks[row] * Ff + col] = f2bf(v > 0.f ? v : 0.f);
        }
      }
  }
  __threadfence();   // feat stores visible block-wide before decode reads
  __syncthreads();

  // ---------------- decode: r = feat @ dec[e] ----------------
  const unsigned short* D0 = decT + (size_t)e * Hh * Ff;
  const unsigned short* f0row = featbuf + (size_t)toks[ln] * Ff + kh * 16;
  const unsigned short* f1row = featbuf + (size_t)toks[16 + ln] * Ff + kh * 16;
#pragma unroll 1
  for (int cg = 0; cg < 2; cg++) {
    int colb = cg * 512 + wave * 64;
    v8f acc[2][4];
#pragma unroll
    for (int rt = 0; rt < 2; rt++)
#pragma unroll
      for (int t = 0; t < 4; t++) { v8f z; for (int j = 0; j < 8; j++) z[j] = 0.f; acc[rt][t] = z; }

    wait_async0();
#pragma unroll
    for (int t = 0; t < 4; t++) {
      const unsigned short* g = D0 + (((size_t)(colb + t * 16 + ln)) << 10) + kh * 16;
      async_b128(bslot + t * 1024u, g);
      async_b128(bslot + t * 1024u + 16u, g + 8);
    }
    unsigned p = 0;
#pragma unroll 1
    for (int k = 0; k < Ff; k += 32) {
      int kn = (k + 32 < Ff) ? k + 32 : k;
      unsigned pn = p ^ 1u;
#pragma unroll
      for (int t = 0; t < 4; t++) {
        const unsigned short* g = D0 + (((size_t)(colb + t * 16 + ln)) << 10) + kn + kh * 16;
        async_b128(bslot + pn * 4096u + t * 1024u, g);
        async_b128(bslot + pn * 4096u + t * 1024u + 16u, g + 8);
      }
      wait_async8();
      Frag a0, a1;
      a0.q[0] = *(const uint4*)(f0row + k);
      a0.q[1] = *(const uint4*)(f0row + k + 8);
      a1.q[0] = *(const uint4*)(f1row + k);
      a1.q[1] = *(const uint4*)(f1row + k + 8);
#pragma unroll
      for (int t = 0; t < 4; t++) {
        Frag b;
        const unsigned char* bp = bread + p * 4096u + t * 1024u;
        b.q[0] = *(const uint4*)(bp);
        b.q[1] = *(const uint4*)(bp + 16);
        acc[0][t] = __builtin_amdgcn_wmma_f32_16x16x32_bf16(
            false, a0.v, false, b.v, (short)0, acc[0][t], false, false);
        acc[1][t] = __builtin_amdgcn_wmma_f32_16x16x32_bf16(
            false, a1.v, false, b.v, (short)0, acc[1][t], false, false);
      }
      p = pn;
    }
    // gated combine + bias, scatter valid rows
#pragma unroll
    for (int rt = 0; rt < 2; rt++)
#pragma unroll
      for (int t = 0; t < 4; t++) {
        int col = colb + t * 16 + ln;
        float bcol = expert_b[col];
#pragma unroll
        for (int j = 0; j < 8; j++) {
          int row = rt * 16 + j + kh * 8;
          if (vld[row])
            out[(size_t)toks[row] * Hh + col] = acc[rt][t][j] * mps[row] + bcol;
        }
      }
  }
}

extern "C" void kernel_launch(void* const* d_in, const int* in_sizes, int n_in,
                              void* d_out, int out_size, void* d_ws, size_t ws_size,
                              hipStream_t stream) {
  (void)in_sizes; (void)n_in; (void)out_size; (void)ws_size;
  const float* act      = (const float*)d_in[0];
  const float* router   = (const float*)d_in[1];
  const float* router_b = (const float*)d_in[2];
  const float* expert_b = (const float*)d_in[3];
  const float* enc      = (const float*)d_in[4];
  const float* dec      = (const float*)d_in[5];
  float* out = (float*)d_out;

  char* ws = (char*)d_ws;
  int*            counts     = (int*)(ws);                        // 64 B (pad 256)
  int*            token_list = (int*)(ws + 256);                  // E*N*4 = 256 KB
  float*          max_prob   = (float*)(ws + 256 + Ee * Nn * 4);  // 16 KB
  unsigned short* encT = (unsigned short*)(ws + 256 + Ee * Nn * 4 + Nn * 4);
  unsigned short* decT = encT + (size_t)Ee * Ff * Hh;             // 32 MB each
  unsigned short* featbuf = decT + (size_t)Ee * Ff * Hh;          // 8 MB

  hipMemsetAsync(counts, 0, Ee * sizeof(int), stream);

  // enc [E,H,F] -> encT [E,F,H] bf16 ; dec [E,F,H] -> decT [E,H,F] bf16
  dim3 tb(32, 8);
  k_transpose<<<dim3(Ff / 32, Hh / 32, Ee), tb, 0, stream>>>(enc, encT, Hh, Ff);
  k_transpose<<<dim3(Hh / 32, Ff / 32, Ee), tb, 0, stream>>>(dec, decT, Ff, Hh);

  k_router<<<Nn / 64, 256, 0, stream>>>(act, router, router_b,
                                        counts, token_list, max_prob);

  k_moe<<<dim3(Nn / 32, Ee), 256, 0, stream>>>(act, expert_b, encT, decT, featbuf,
                                               counts, token_list, max_prob, out);
}
